// Net_65249143160876
// MI455X (gfx1250) — matrix operand order repaired and checked by reading, not compiled, not run on hardware
//
#include <hip/hip_runtime.h>

#define BB   128
#define NN   1000
#define KNN  15
#define LL   7
#define FNR  10
#define KSP  5
#define HID  256
#define CC   40

typedef __attribute__((ext_vector_type(16))) _Float16 v16h;
typedef __attribute__((ext_vector_type(8)))  float    v8f;

__device__ __forceinline__ void sym3_sq(float& xx, float& xy, float& xz,
                                        float& yy, float& yz, float& zz) {
    float rxx = xx*xx + xy*xy + xz*xz;
    float rxy = xx*xy + xy*yy + xz*yz;
    float rxz = xx*xz + xy*yz + xz*zz;
    float ryy = xy*xy + yy*yy + yz*yz;
    float ryz = xy*xz + yy*yz + yz*zz;
    float rzz = xz*xz + yz*yz + zz*zz;
    xx = rxx; xy = rxy; xz = rxz; yy = ryy; yz = ryz; zz = rzz;
}

// ---------------------------------------------------------------------------
// Kernel 1: per-batch KNN + directional spline features + sigmoid + mean
// One workgroup (256 threads = 8 waves) per batch; 1000 points LDS-resident.
// ---------------------------------------------------------------------------
__global__ __launch_bounds__(256) void k1_knn_feat(const float* __restrict__ pos,
                                                   const float* __restrict__ w_spline,
                                                   float* __restrict__ ys) {
    __shared__ float px[NN], py[NN], pz[NN], sq[NN];
    __shared__ float wsp[FNR * KSP];
    __shared__ float red[256][FNR];

    const int b   = blockIdx.x;
    const int tid = threadIdx.x;
    const float* pb = pos + (size_t)b * NN * 3;

    if (tid == 0) __builtin_prefetch(pb, 0, 1);   // global_prefetch_b8

    for (int i = tid; i < NN; i += 256) {
        float x = pb[i*3+0], y = pb[i*3+1], z = pb[i*3+2];
        px[i] = x; py[i] = y; pz[i] = z; sq[i] = x*x + y*y + z*z;
    }
    if (tid < FNR * KSP) wsp[tid] = w_spline[tid];

    float acc[FNR];
    #pragma unroll
    for (int f = 0; f < FNR; ++f) acc[f] = 0.f;
    __syncthreads();

    for (int i = tid; i < NN; i += 256) {
        const float qx = px[i], qy = py[i], qz = pz[i], qs = sq[i];

        // sorted top-KNN (ascending d2), static indexing -> stays in VGPRs
        float bd[KNN]; int bj[KNN];
        #pragma unroll
        for (int k = 0; k < KNN; ++k) { bd[k] = 3.0e38f; bj[k] = 0; }

        for (int j = 0; j < NN; ++j) {
            float d2 = qs + sq[j] - 2.0f * (qx*px[j] + qy*py[j] + qz*pz[j]);
            if (j == i) d2 += 1.0e10f;           // matches eye()*1e10
            if (d2 < bd[KNN-1]) {
                bd[KNN-1] = d2; bj[KNN-1] = j;
                #pragma unroll
                for (int s = KNN-1; s > 0; --s) {
                    if (bd[s] < bd[s-1]) {
                        float td = bd[s]; bd[s] = bd[s-1]; bd[s-1] = td;
                        int   tj = bj[s]; bj[s] = bj[s-1]; bj[s-1] = tj;
                    }
                }
            }
        }

        float rx[KNN], ry[KNN], rz[KNN];
        #pragma unroll
        for (int k = 0; k < KNN; ++k) {
            int j = bj[k];
            rx[k] = px[j] - qx; ry[k] = py[j] - qy; rz[k] = pz[j] - qz;
        }

        // covariance of the 7 nearest
        float cxx = 0, cxy = 0, cxz = 0, cyy = 0, cyz = 0, czz = 0;
        #pragma unroll
        for (int k = 0; k < LL; ++k) {
            cxx += rx[k]*rx[k]; cxy += rx[k]*ry[k]; cxz += rx[k]*rz[k];
            cyy += ry[k]*ry[k]; cyz += ry[k]*rz[k]; czz += rz[k]*rz[k];
        }

        // principal eigenvector: normalize, square 3x (C^8), power-iterate
        float invtr = 1.0f / (cxx + cyy + czz + 1e-30f);
        float axx = cxx*invtr, axy = cxy*invtr, axz = cxz*invtr;
        float ayy = cyy*invtr, ayz = cyz*invtr, azz = czz*invtr;
        sym3_sq(axx, axy, axz, ayy, ayz, azz);   // C^2
        sym3_sq(axx, axy, axz, ayy, ayz, azz);   // C^4
        sym3_sq(axx, axy, axz, ayy, ayz, azz);   // C^8
        float vx = 0.57735f, vy = 0.57735f, vz = 0.57735f;
        #pragma unroll
        for (int it = 0; it < 6; ++it) {
            float tx = axx*vx + axy*vy + axz*vz;
            float ty = axy*vx + ayy*vy + ayz*vz;
            float tz = axz*vx + ayz*vy + azz*vz;
            float inv = rsqrtf(tx*tx + ty*ty + tz*tz + 1e-30f);
            vx = tx*inv; vy = ty*inv; vz = tz*inv;
        }
        if (vx < 0.f) { vx = -vx; vy = -vy; vz = -vz; }   // sign convention

        float rmax = 0.f;
        #pragma unroll
        for (int k = 0; k < KNN; ++k)
            rmax = fmaxf(rmax, rx[k]*rx[k] + ry[k]*ry[k] + rz[k]*rz[k]);
        float rinv = 1.0f / (sqrtf(rmax) + 1e-8f);

        // accumulate hat-basis sums over neighbors
        float bs[KSP];
        #pragma unroll
        for (int m = 0; m < KSP; ++m) bs[m] = 0.f;
        #pragma unroll
        for (int k = 0; k < KNN; ++k) {
            float t = (rx[k]*vx + ry[k]*vy + rz[k]*vz) * rinv;
            #pragma unroll
            for (int m = 0; m < KSP; ++m) {
                float knot = -1.0f + 0.5f * (float)m;
                bs[m] += fmaxf(0.f, 1.0f - fabsf(t - knot) * 2.0f);
            }
        }
        #pragma unroll
        for (int f = 0; f < FNR; ++f) {
            float feat = 0.f;
            #pragma unroll
            for (int m = 0; m < KSP; ++m) feat += bs[m] * wsp[f*KSP + m];
            feat *= (1.0f / (float)KNN);
            acc[f] += 1.0f / (1.0f + __expf(-feat));     // sigmoid
        }
    }

    #pragma unroll
    for (int f = 0; f < FNR; ++f) red[tid][f] = acc[f];
    __syncthreads();
    for (int s = 128; s > 0; s >>= 1) {
        if (tid < s) {
            #pragma unroll
            for (int f = 0; f < FNR; ++f) red[tid][f] += red[tid + s][f];
        }
        __syncthreads();
    }
    if (tid < FNR) ys[b*FNR + tid] = red[0][tid] * (1.0f / (float)NN);
}

// ---------------------------------------------------------------------------
// Kernel 2: y1 = elu(ys @ w1 + b1) via v_wmma_f32_16x16x32_f16 (K=10 padded)
// One workgroup, 8 waves x 16-row strips x 16 N-tiles.
// ---------------------------------------------------------------------------
__global__ __launch_bounds__(256) void k2_fc1(const float* __restrict__ ys,
                                              const float* __restrict__ w1,
                                              const float* __restrict__ b1,
                                              _Float16* __restrict__ y1h) {
    __shared__ _Float16 sy[BB * 32];     // 128 x 32, K zero-padded
    __shared__ _Float16 sw[32 * HID];    // 32 x 256, K zero-padded
    const int tid = threadIdx.x;

    for (int idx = tid; idx < BB * 32; idx += 256) {
        int m = idx >> 5, k = idx & 31;
        sy[idx] = (k < FNR) ? (_Float16)ys[m*FNR + k] : (_Float16)0.f;
    }
    for (int idx = tid; idx < 32 * HID; idx += 256) {
        int k = idx >> 8, n = idx & 255;
        sw[idx] = (k < FNR) ? (_Float16)w1[k*HID + n] : (_Float16)0.f;
    }
    __syncthreads();

    const int wave = tid >> 5, lane = tid & 31;
    const int base = wave * 16;
    const int lh   = (lane >= 16) ? 1 : 0;
    const int lm   = lane & 15;

    // A fragment (16x32 f16, ISA 7.12.2 layout) — invariant across N-tiles
    v16h a;
    #pragma unroll
    for (int e = 0; e < 16; ++e) {
        int K = e + ((e < 8) ? (lh ? 8 : 0) : (lh ? 16 : 8));
        a[e] = sy[(base + lm) * 32 + K];
    }

    for (int nt = 0; nt < HID / 16; ++nt) {
        v16h bf;
        #pragma unroll
        for (int j = 0; j < 16; ++j) {
            int K = j + ((j < 8) ? (lh ? 8 : 0) : (lh ? 16 : 8));
            bf[j] = sw[K*HID + nt*16 + lm];
        }
        v8f c = {};
        c = __builtin_amdgcn_wmma_f32_16x16x32_f16(false, a, false, bf,
                                                   (short)0, c, false, false);
        #pragma unroll
        for (int r = 0; r < 8; ++r) {
            int M = base + (lh ? 8 + r : r);
            int N = nt*16 + lm;
            float v = c[r] + b1[N];
            v = (v > 0.f) ? v : (__expf(v) - 1.0f);       // elu
            y1h[M*HID + N] = (_Float16)v;
        }
    }
}

// ---------------------------------------------------------------------------
// Kernel 3: y2 = y1 @ w2 + b2 (WMMA, 8 K-chunks x 3 N-tiles), then log_softmax
// ---------------------------------------------------------------------------
__global__ __launch_bounds__(256) void k3_fc2_lsm(const _Float16* __restrict__ y1h,
                                                  const float* __restrict__ w2,
                                                  const float* __restrict__ b2,
                                                  float* __restrict__ out) {
    __shared__ _Float16 sw[HID * 48];    // 256 x 48, N zero-padded
    __shared__ float    y2[BB * 48];
    const int tid = threadIdx.x;

    for (int idx = tid; idx < HID * 48; idx += 256) {
        int k = idx / 48, n = idx % 48;
        sw[idx] = (n < CC) ? (_Float16)w2[k*CC + n] : (_Float16)0.f;
    }
    __syncthreads();

    const int wave = tid >> 5, lane = tid & 31;
    const int base = wave * 16;
    const int lh   = (lane >= 16) ? 1 : 0;
    const int lm   = lane & 15;

    v8f c0 = {}, c1 = {}, c2 = {};
    for (int kc = 0; kc < HID / 32; ++kc) {
        const int kb = kc * 32;
        v16h a;
        #pragma unroll
        for (int e = 0; e < 16; ++e) {
            int K = kb + e + ((e < 8) ? (lh ? 8 : 0) : (lh ? 16 : 8));
            a[e] = y1h[(base + lm) * HID + K];
        }
        v16h b0f, b1f, b2f;
        #pragma unroll
        for (int j = 0; j < 16; ++j) {
            int K = kb + j + ((j < 8) ? (lh ? 8 : 0) : (lh ? 16 : 8));
            b0f[j] = sw[K*48 +  0 + lm];
            b1f[j] = sw[K*48 + 16 + lm];
            b2f[j] = sw[K*48 + 32 + lm];
        }
        c0 = __builtin_amdgcn_wmma_f32_16x16x32_f16(false, a, false, b0f, (short)0, c0, false, false);
        c1 = __builtin_amdgcn_wmma_f32_16x16x32_f16(false, a, false, b1f, (short)0, c1, false, false);
        c2 = __builtin_amdgcn_wmma_f32_16x16x32_f16(false, a, false, b2f, (short)0, c2, false, false);
    }

    #pragma unroll
    for (int r = 0; r < 8; ++r) {
        int M = base + (lh ? 8 + r : r);
        int col;
        col =  0 + lm; y2[M*48 + col] = c0[r] + ((col < CC) ? b2[col] : 0.f);
        col = 16 + lm; y2[M*48 + col] = c1[r] + ((col < CC) ? b2[col] : 0.f);
        col = 32 + lm; y2[M*48 + col] = c2[r] + ((col < CC) ? b2[col] : 0.f);
    }
    __syncthreads();

    if (tid < BB) {
        float mx = -3e38f;
        for (int c = 0; c < CC; ++c) mx = fmaxf(mx, y2[tid*48 + c]);
        float s = 0.f;
        for (int c = 0; c < CC; ++c) s += __expf(y2[tid*48 + c] - mx);
        float lse = mx + __logf(s);
        for (int c = 0; c < CC; ++c) out[tid*CC + c] = y2[tid*48 + c] - lse;
    }
}

// ---------------------------------------------------------------------------
extern "C" void kernel_launch(void* const* d_in, const int* in_sizes, int n_in,
                              void* d_out, int out_size, void* d_ws, size_t ws_size,
                              hipStream_t stream) {
    const float* pos      = (const float*)d_in[0];
    // d_in[1] = batch (unused by reference)
    const float* w_spline = (const float*)d_in[2];
    const float* w1       = (const float*)d_in[3];
    const float* b1       = (const float*)d_in[4];
    const float* w2       = (const float*)d_in[5];
    const float* b2       = (const float*)d_in[6];

    float*    ys  = (float*)d_ws;                               // 128*10 f32
    _Float16* y1h = (_Float16*)((char*)d_ws + 8192);            // 128*256 f16

    k1_knn_feat<<<BB, 256, 0, stream>>>(pos, w_spline, ys);
    k2_fc1     <<<1,  256, 0, stream>>>(ys, w1, b1, y1h);
    k3_fc2_lsm <<<1,  256, 0, stream>>>(y1h, w2, b2, (float*)d_out);
}
